// Relations_257698038035
// MI455X (gfx1250) — compile-verified, matrix operand order checked by reading, not run on hardware
//
#include <hip/hip_runtime.h>
#include <hip/hip_bf16.h>

// ---------------------------------------------------------------------------
// Relations module (mmdet-style ROI attention) for MI455X / gfx1250.
// All heavy math on v_wmma_f32_16x16x32_f16 (wave32 WMMA, fp32 accumulate).
// Flash-style fusion: (N,G,N) score/attn tensors never materialized.
// in_x is pre-transposed once to f16 (C x N) so PV B-fragments stream
// K-contiguous straight from L2 (192 MB -> whole working set resident).
// ---------------------------------------------------------------------------

#define NROI 2048
#define CDIM 1024
#define NGRP 16
#define DDIM 64

typedef __attribute__((ext_vector_type(16))) _Float16 v16h;
typedef __attribute__((ext_vector_type(8)))  float    v8f;
typedef int vint4 __attribute__((vector_size(16)));

#define WMMA_F16(a, b, c) \
    __builtin_amdgcn_wmma_f32_16x16x32_f16(false, (a), false, (b), (short)0, (c), false, false)

#if __has_builtin(__builtin_amdgcn_global_load_async_to_lds_b128) && \
    __has_builtin(__builtin_amdgcn_s_wait_asynccnt)
#define USE_ASYNC_LDS 1
#else
#define USE_ASYNC_LDS 0
#endif

// 16-byte global -> LDS copy; async (ASYNCcnt-tracked) when available.
__device__ __forceinline__ void copy16_g2l(const _Float16* g, _Float16* l) {
#if USE_ASYNC_LDS
    __builtin_amdgcn_global_load_async_to_lds_b128(
        (__attribute__((address_space(1))) vint4*)(g),
        (__attribute__((address_space(3))) vint4*)(l), 0, 0);
#else
    uint4 v = *(const uint4*)g;
    *(uint4*)l = v;
#endif
}

__device__ __forceinline__ void wait_staging() {
#if USE_ASYNC_LDS
    __builtin_amdgcn_s_wait_asynccnt(0);
#endif
}

// Load a 16x32 f16 fragment whose K-run is contiguous per lane-row.
// rowp points at [laneRow][k=0]; rows must be 16B aligned so the paired dword
// loads coalesce into ds_load_b128 / global_load_b128.
// CDNA5 16-bit operand striping: VGPR r, lanes 0-15: k = kbase(r),
// lanes 16-31: k = kbase(r)+8; kbase(r) = 2r (r<4), 16+2(r-4) (r>=4).
__device__ __forceinline__ v16h frag_load(const _Float16* rowp) {
    int hf = (threadIdx.x >> 4) & 1;
    union { v16h v; unsigned u[8]; } f;
#pragma unroll
    for (int r = 0; r < 8; ++r) {
        int k = (r < 4 ? 2 * r : 16 + 2 * (r - 4)) + 8 * hf;
        f.u[r] = *(const unsigned*)(rowp + k);
    }
    return f.v;
}

// ---------------------------------------------------------------------------
// elementwise converts
// ---------------------------------------------------------------------------
__global__ __launch_bounds__(256) void cvt_f16_kernel(const float* __restrict__ in,
                                                      _Float16* __restrict__ out, int n) {
    int i = blockIdx.x * 256 + threadIdx.x;
    if (i < n) out[i] = (_Float16)in[i];
}

__global__ __launch_bounds__(256) void relu_cvt_kernel(const float* __restrict__ in,
                                                       _Float16* __restrict__ out, int n) {
    int i = blockIdx.x * 256 + threadIdx.x;
    if (i < n) out[i] = (_Float16)fmaxf(in[i], 0.0f);
}

// in (N x C f32, row-major) -> outT (C x N f16, row-major) via 32x32 LDS tiles
__global__ __launch_bounds__(256) void transpose_f16_kernel(const float* __restrict__ in,
                                                            _Float16* __restrict__ outT) {
    __shared__ _Float16 sT[32][33];
    const int m0 = blockIdx.x * 32;
    const int c0 = blockIdx.y * 32;
    const int tr = threadIdx.x >> 5;   // 0..7
    const int tc = threadIdx.x & 31;
#pragma unroll
    for (int rep = 0; rep < 4; ++rep) {
        int row = tr + rep * 8;
        sT[tc][row] = (_Float16)in[(size_t)(m0 + row) * CDIM + c0 + tc];
    }
    __syncthreads();
#pragma unroll
    for (int rep = 0; rep < 4; ++rep) {
        int cr = tr + rep * 8;
        outT[(size_t)(c0 + cr) * NROI + m0 + tc] = sT[cr][tc];
    }
}

// ---------------------------------------------------------------------------
// L[n,m] = log(IoU(roi_n, roi_m) + 1e-6), mmdet +1 pixel convention
// ---------------------------------------------------------------------------
__global__ __launch_bounds__(256) void iou_log_kernel(const float* __restrict__ rois,
                                                      float* __restrict__ L) {
    size_t t = (size_t)blockIdx.x * 256 + threadIdx.x;
    int n = (int)(t >> 11);
    int m = (int)(t & 2047);
    const float* a = rois + (size_t)n * 5 + 1;
    const float* b = rois + (size_t)m * 5 + 1;
    float ax1 = a[0], ay1 = a[1], ax2 = a[2], ay2 = a[3];
    float bx1 = b[0], by1 = b[1], bx2 = b[2], by2 = b[3];
    float areaA = (ax2 - ax1 + 1.0f) * (ay2 - ay1 + 1.0f);
    float areaB = (bx2 - bx1 + 1.0f) * (by2 - by1 + 1.0f);
    float w = fmaxf(fminf(ax2, bx2) - fmaxf(ax1, bx1) + 1.0f, 0.0f);
    float h = fmaxf(fminf(ay2, by2) - fmaxf(ay1, by1) + 1.0f, 0.0f);
    float inter = w * h;
    float iou = inter / (areaA + areaB - inter);
    L[t] = __logf(iou + 1e-6f);
}

// ---------------------------------------------------------------------------
// out[m,c] = (f16)( maybe_relu( sum_k A[m,k]*B[c,k] + bias[c] ) )
// A: 2048x1024 f16 (already ReLU'd), B: 1024x1024 f16.  128x128 block tiles,
// 8 waves = 4(M) x 2(N); K staged 32 at a time through LDS (async copies).
// LDS row stride 40 halves -> conflict-free ds_load_b128 fragment reads.
// ---------------------------------------------------------------------------
#define LDE 40

template <bool RELU_OUT>
__global__ __launch_bounds__(256) void gemm_f16_kernel(const _Float16* __restrict__ A,
                                                       const _Float16* __restrict__ B,
                                                       const float* __restrict__ bias,
                                                       _Float16* __restrict__ out) {
    __shared__ _Float16 sA[128 * LDE];
    __shared__ _Float16 sB[128 * LDE];

    const int tid  = threadIdx.x;
    const int wid  = tid >> 5;
    const int lane = tid & 31;
    const int idx  = lane & 15;
    const int hf   = lane >> 4;
    const int m0 = blockIdx.x * 128;
    const int c0 = blockIdx.y * 128;
    const int waveM = wid & 3;
    const int waveN = wid >> 2;

    v8f acc[2][4] = {};

    const int srow = tid >> 1;        // 0..127
    const int skp  = (tid & 1) * 16;  // 0 or 16

    for (int k0 = 0; k0 < CDIM; k0 += 32) {
        const _Float16* ga = A + (size_t)(m0 + srow) * CDIM + k0 + skp;
        const _Float16* gb = B + (size_t)(c0 + srow) * CDIM + k0 + skp;
        _Float16* la = &sA[srow * LDE + skp];
        _Float16* lb = &sB[srow * LDE + skp];
        copy16_g2l(ga, la);
        copy16_g2l(ga + 8, la + 8);
        copy16_g2l(gb, lb);
        copy16_g2l(gb + 8, lb + 8);
        wait_staging();
        __syncthreads();

        v16h af[2], bf[4];
#pragma unroll
        for (int sm = 0; sm < 2; ++sm)
            af[sm] = frag_load(&sA[(waveM * 32 + sm * 16 + idx) * LDE]);
#pragma unroll
        for (int sn = 0; sn < 4; ++sn)
            bf[sn] = frag_load(&sB[(waveN * 64 + sn * 16 + idx) * LDE]);
#pragma unroll
        for (int sm = 0; sm < 2; ++sm)
#pragma unroll
            for (int sn = 0; sn < 4; ++sn)
                acc[sm][sn] = WMMA_F16(af[sm], bf[sn], acc[sm][sn]);
        __syncthreads();
    }

#pragma unroll
    for (int sm = 0; sm < 2; ++sm) {
#pragma unroll
        for (int sn = 0; sn < 4; ++sn) {
            union { v8f v; float f[8]; } u;
            u.v = acc[sm][sn];
            int c = c0 + waveN * 64 + sn * 16 + idx;
            float bv = bias[c];
#pragma unroll
            for (int r = 0; r < 8; ++r) {
                int m = m0 + waveM * 32 + sm * 16 + r + 8 * hf;
                float v = u.f[r] + bv;
                if (RELU_OUT) v = fmaxf(v, 0.0f);
                out[(size_t)m * CDIM + c] = (_Float16)v;
            }
        }
    }
}

// ---------------------------------------------------------------------------
// Pass 1: per (n, g) softmax stats over m.  s = (Xg[n]·Xg[m])/8 + L[n,m].
// ---------------------------------------------------------------------------
__global__ __launch_bounds__(256) void stats_kernel(const _Float16* __restrict__ Xh,
                                                    const float* __restrict__ L,
                                                    float* __restrict__ rmax,
                                                    float* __restrict__ rsum) {
    const int tid  = threadIdx.x;
    const int wid  = tid >> 5;
    const int lane = tid & 31;
    const int idx  = lane & 15;
    const int hf   = lane >> 4;
    const int n0 = blockIdx.x * 16;
    const int g  = blockIdx.y;

    const _Float16* pa = Xh + (size_t)(n0 + idx) * CDIM + g * DDIM;
    v16h a0 = frag_load(pa);
    v16h a1 = frag_load(pa + 32);

    float M8[8], Z8[8];
#pragma unroll
    for (int r = 0; r < 8; ++r) { M8[r] = -1e30f; Z8[r] = 0.0f; }

    for (int t = wid; t < NROI / 16; t += 8) {
        int mc0 = t * 16;
        const _Float16* pb = Xh + (size_t)(mc0 + idx) * CDIM + g * DDIM;
        v16h b0 = frag_load(pb);
        v16h b1 = frag_load(pb + 32);
        v8f s = {};
        s = WMMA_F16(a0, b0, s);
        s = WMMA_F16(a1, b1, s);
        union { v8f v; float f[8]; } u;
        u.v = s;
#pragma unroll
        for (int r = 0; r < 8; ++r) {
            int n = n0 + r + 8 * hf;
            float sv = u.f[r] * 0.125f + L[(size_t)n * NROI + mc0 + idx];
            float nm = fmaxf(M8[r], sv);
            Z8[r] = Z8[r] * __expf(M8[r] - nm) + __expf(sv - nm);
            M8[r] = nm;
        }
    }

#pragma unroll
    for (int r = 0; r < 8; ++r) {
#pragma unroll
        for (int mask = 1; mask < 16; mask <<= 1) {
            float oM = __shfl_xor(M8[r], mask, 32);
            float oZ = __shfl_xor(Z8[r], mask, 32);
            float nm = fmaxf(M8[r], oM);
            Z8[r] = Z8[r] * __expf(M8[r] - nm) + oZ * __expf(oM - nm);
            M8[r] = nm;
        }
    }

    __shared__ float sM[8][16];
    __shared__ float sZ[8][16];
    if (idx == 0) {
#pragma unroll
        for (int r = 0; r < 8; ++r) {
            sM[wid][r + 8 * hf] = M8[r];
            sZ[wid][r + 8 * hf] = Z8[r];
        }
    }
    __syncthreads();
    if (tid < 16) {
        float m = sM[0][tid], z = sZ[0][tid];
#pragma unroll
        for (int w = 1; w < 8; ++w) {
            float oM = sM[w][tid], oZ = sZ[w][tid];
            float nm = fmaxf(m, oM);
            z = z * __expf(m - nm) + oZ * __expf(oM - nm);
            m = nm;
        }
        rmax[(size_t)(n0 + tid) * NGRP + g] = m;
        rsum[(size_t)(n0 + tid) * NGRP + g] = z;
    }
}

// ---------------------------------------------------------------------------
// Pass 2: y[n,g,:] = softmax_m(s) @ in_x, fused grouped projection:
// out[n, g*64+o] = sum_c y[n,g,c] * Wout[g,o,c] + bout[g*64+o].
// m tiled by 128 (all 8 waves compute S); PV B-frags stream K-contiguous
// from the pre-transposed in_xT (C x N f16) straight out of L2.
// ---------------------------------------------------------------------------
#define LDP 136   // sP row stride (halves): DWORD stride 68 -> conflict-free
#define LDY 1032  // sY row stride (halves): DWORD stride 516 -> conflict-free

__global__ __launch_bounds__(256) void pv_kernel(const _Float16* __restrict__ Xh,
                                                 const _Float16* __restrict__ inXT,
                                                 const _Float16* __restrict__ WoutH,
                                                 const float* __restrict__ L,
                                                 const float* __restrict__ rmax,
                                                 const float* __restrict__ rsum,
                                                 const float* __restrict__ bout,
                                                 float* __restrict__ out) {
    __shared__ _Float16 sP[16 * LDP];
    __shared__ _Float16 sY[16 * LDY];
    __shared__ float    sOut[16 * 64];

    const int tid  = threadIdx.x;
    const int wid  = tid >> 5;
    const int lane = tid & 31;
    const int idx  = lane & 15;
    const int hf   = lane >> 4;
    const int n0 = blockIdx.x * 16;
    const int g  = blockIdx.y;

    const _Float16* pa = Xh + (size_t)(n0 + idx) * CDIM + g * DDIM;
    v16h a0 = frag_load(pa);
    v16h a1 = frag_load(pa + 32);

    float M8[8], Zi8[8];
#pragma unroll
    for (int r = 0; r < 8; ++r) {
        int n = n0 + r + 8 * hf;
        M8[r]  = rmax[(size_t)n * NGRP + g];
        Zi8[r] = 1.0f / rsum[(size_t)n * NGRP + g];
    }

    for (int i = tid; i < 16 * 64; i += 256) sOut[i] = 0.0f;

    v8f acc[8] = {};
    const int c0w = wid * 128;

    for (int m0 = 0; m0 < NROI; m0 += 128) {
        // every wave computes one 16x16 S subtile and writes normalized P
        int mc0 = m0 + wid * 16;
        const _Float16* pb = Xh + (size_t)(mc0 + idx) * CDIM + g * DDIM;
        v16h b0 = frag_load(pb);
        v16h b1 = frag_load(pb + 32);
        v8f s = {};
        s = WMMA_F16(a0, b0, s);
        s = WMMA_F16(a1, b1, s);
        union { v8f v; float f[8]; } u;
        u.v = s;
#pragma unroll
        for (int r = 0; r < 8; ++r) {
            int n = n0 + r + 8 * hf;
            float p = __expf(u.f[r] * 0.125f + L[(size_t)n * NROI + mc0 + idx] - M8[r]) * Zi8[r];
            sP[(r + 8 * hf) * LDP + wid * 16 + idx] = (_Float16)p;
        }
        __syncthreads();

        // PV: acc(16 x 128-chunk) += P(16x128) @ in_x(128 x 128-chunk)
        const _Float16* pp = sP + idx * LDP;
        v16h pf[4];
#pragma unroll
        for (int kk = 0; kk < 4; ++kk) pf[kk] = frag_load(pp + kk * 32);
#pragma unroll
        for (int j = 0; j < 8; ++j) {
            const _Float16* pv = inXT + (size_t)(c0w + j * 16 + idx) * NROI + m0;
            __builtin_prefetch(pv + 128, 0, 1);  // next m-tile row -> L2
#pragma unroll
            for (int kk = 0; kk < 4; ++kk) {
                v16h bf = frag_load(pv + kk * 32);
                acc[j] = WMMA_F16(pf[kk], bf, acc[j]);
            }
        }
        __syncthreads();
    }

    // spill y (16 x 1024) to LDS as f16, K-contiguous per row for A frags
#pragma unroll
    for (int j = 0; j < 8; ++j) {
        union { v8f v; float f[8]; } u;
        u.v = acc[j];
#pragma unroll
        for (int r = 0; r < 8; ++r)
            sY[(size_t)(r + 8 * hf) * LDY + c0w + j * 16 + idx] = (_Float16)u.f[r];
    }
    __syncthreads();

    // fused grouped projection: out16x64 += y16x128chunk @ Wout[g,:,chunk]^T
    const int kc = wid * 128;
    v8f oacc[4] = {};
    const _Float16* pya = sY + (size_t)idx * LDY + kc;
#pragma unroll
    for (int kk = 0; kk < 4; ++kk) {
        v16h af = frag_load(pya + kk * 32);
#pragma unroll
        for (int j = 0; j < 4; ++j) {
            const _Float16* pw = WoutH + (size_t)(g * DDIM + j * 16 + idx) * CDIM + kc + kk * 32;
            v16h bf = frag_load(pw);
            oacc[j] = WMMA_F16(af, bf, oacc[j]);
        }
    }
#pragma unroll
    for (int j = 0; j < 4; ++j) {
        union { v8f v; float f[8]; } u;
        u.v = oacc[j];
#pragma unroll
        for (int r = 0; r < 8; ++r)
            atomicAdd(&sOut[(r + 8 * hf) * 64 + j * 16 + idx], u.f[r]);
    }
    __syncthreads();

    for (int i = tid; i < 16 * 64; i += 256) {
        int row = i >> 6, o = i & 63;
        out[(size_t)(n0 + row) * CDIM + g * DDIM + o] = sOut[i] + bout[g * DDIM + o];
    }
}

// ---------------------------------------------------------------------------
// Launch
// ---------------------------------------------------------------------------
extern "C" void kernel_launch(void* const* d_in, const int* in_sizes, int n_in,
                              void* d_out, int out_size, void* d_ws, size_t ws_size,
                              hipStream_t stream) {
    const float* in_x = (const float*)d_in[0];
    const float* rois = (const float*)d_in[1];
    const float* W0   = (const float*)d_in[2];
    const float* b0   = (const float*)d_in[3];
    const float* W1   = (const float*)d_in[4];
    const float* b1   = (const float*)d_in[5];
    const float* Wout = (const float*)d_in[6];
    const float* bout = (const float*)d_in[7];
    float* out = (float*)d_out;

    char* ws = (char*)d_ws;
    _Float16* X0h   = (_Float16*)(ws);                        // 4 MB  relu(in_x) f16
    _Float16* X1h   = (_Float16*)(ws + ((size_t)4  << 20));   // 4 MB  relu(embed1) f16
    _Float16* Xh    = (_Float16*)(ws + ((size_t)8  << 20));   // 4 MB  embed2 f16
    _Float16* W0h   = (_Float16*)(ws + ((size_t)12 << 20));   // 2 MB
    _Float16* W1h   = (_Float16*)(ws + ((size_t)14 << 20));   // 2 MB
    _Float16* WoutH = (_Float16*)(ws + ((size_t)16 << 20));   // 2 MB
    _Float16* inXT  = (_Float16*)(ws + ((size_t)18 << 20));   // 4 MB  in_x^T f16
    float*    Lbuf  = (float*)(ws + ((size_t)22 << 20));      // 16 MB log-IoU
    float*    rmax  = (float*)(ws + ((size_t)38 << 20));      // 128 KB
    float*    rsum  = (float*)(ws + ((size_t)38 << 20) + (256u << 10));

    const int nc = NROI * CDIM;
    relu_cvt_kernel<<<nc / 256, 256, 0, stream>>>(in_x, X0h, nc);
    cvt_f16_kernel<<<(CDIM * CDIM) / 256, 256, 0, stream>>>(W0, W0h, CDIM * CDIM);
    cvt_f16_kernel<<<(CDIM * CDIM) / 256, 256, 0, stream>>>(W1, W1h, CDIM * CDIM);
    cvt_f16_kernel<<<(NGRP * DDIM * CDIM) / 256, 256, 0, stream>>>(Wout, WoutH,
                                                                   NGRP * DDIM * CDIM);
    transpose_f16_kernel<<<dim3(NROI / 32, CDIM / 32), 256, 0, stream>>>(in_x, inXT);
    iou_log_kernel<<<(NROI * NROI) / 256, 256, 0, stream>>>(rois, Lbuf);

    dim3 eg(NROI / 128, CDIM / 128);
    gemm_f16_kernel<true><<<eg, 256, 0, stream>>>(X0h, W0h, b0, X1h);
    gemm_f16_kernel<false><<<eg, 256, 0, stream>>>(X1h, W1h, b1, Xh);

    dim3 ag(NROI / 16, NGRP);
    stats_kernel<<<ag, 256, 0, stream>>>(Xh, Lbuf, rmax, rsum);
    pv_kernel<<<ag, 256, 0, stream>>>(Xh, inXT, WoutH, Lbuf, rmax, rsum, bout, out);
}
